// SkipLSTM_18416819765878
// MI455X (gfx1250) — compile-verified, matrix operand order checked by reading, not run on hardware
//
#include <hip/hip_runtime.h>

// ---------------- problem constants ----------------
constexpr int T_  = 512;
constexpr int BS  = 256;
constexpr int IC  = 128;
constexpr int HC  = 256;
constexpr int G   = 4 * HC;      // 1024 gate columns

constexpr int GEMM_BLOCKS  = 128;   // 128 blocks x 128 threads = 512 waves
constexpr int GEMM_THREADS = 128;

typedef _Float16 half_t;
typedef __attribute__((ext_vector_type(16))) _Float16 v16h;
typedef __attribute__((ext_vector_type(8)))  _Float16 v8h;
typedef __attribute__((ext_vector_type(8)))  float    v8f;

union HFrag { v16h v; v8h h[2]; };

// Load one WMMA 16x32 f16 fragment row per ISA layout:
// lane = M (or N) row & 15 ; khalf = (lane>>4)*8 selects low/high K half.
// Two aligned 16B loads per lane.
__device__ __forceinline__ v16h load_frag(const half_t* __restrict__ row,
                                          int kBase, int khalf) {
  HFrag f;
  f.h[0] = *(const v8h*)(row + kBase + khalf);
  f.h[1] = *(const v8h*)(row + kBase + 16 + khalf);
  return f.v;
}

__device__ __forceinline__ float sigf(float x) { return 1.0f / (1.0f + expf(-x)); }

// ---------------- grid-wide split barrier (sense via generation counter) ----
__device__ __forceinline__ void grid_barrier(unsigned* __restrict__ cnt,
                                             volatile unsigned* gen,
                                             unsigned nb) {
  __syncthreads();
  if (threadIdx.x == 0) {
    __threadfence();                       // make my writes device-visible
    unsigned g = *gen;
    if (atomicAdd(cnt, 1u) == nb - 1u) {
      *cnt = 0u;                           // safe: everyone has arrived
      __threadfence();
      atomicAdd((unsigned*)gen, 1u);       // release waiters
    } else {
      while (*gen == g) {
        __builtin_amdgcn_s_sleep(1);
      }
    }
    __threadfence();                       // acquire others' writes
  }
  __syncthreads();
}

// ---------------- f32 -> f16 conversion ----------------
__global__ __launch_bounds__(256) void cvt_f32_f16(const float* __restrict__ src,
                                                   half_t* __restrict__ dst, int n) {
  int i = blockIdx.x * blockDim.x + threadIdx.x;
  if (i < n) dst[i] = (half_t)src[i];
}

// ---------------- state init ----------------
__global__ __launch_bounds__(256) void init_state(float* __restrict__ h32,
                                                  half_t* __restrict__ hbf,
                                                  float* __restrict__ c32,
                                                  float* __restrict__ u,
                                                  float* __restrict__ du,
                                                  int* __restrict__ skip,
                                                  unsigned* __restrict__ barCnt,
                                                  unsigned* __restrict__ barGen,
                                                  int resetSkip) {
  int i = blockIdx.x * blockDim.x + threadIdx.x;
  if (i < BS * HC) { h32[i] = 0.0f; hbf[i] = (half_t)0.0f; c32[i] = 0.0f; }
  if (i < BS) {
    u[i] = 1.0f;
    if (resetSkip) { du[i] = 0.0f; skip[i] = 0; }
  }
  if (i == 0 && resetSkip) { *barCnt = 0u; *barGen = 0u; }
}

// ---------------- persistent SkipLSTM layer kernel ----------------
// 512 resident waves. Per timestep:
//   phase 1: gates = X_t @ Wih^T + H @ Whh^T   (wave -> 16x32 output patch)
//   grid barrier
//   phase 2: waves 0..255 run the skip-LSTM cell, one wave32 per batch row
//   grid barrier (publish new h before next step's GEMM)
__global__ __launch_bounds__(GEMM_THREADS) void skiplstm_layer(
    const half_t* __restrict__ Xall, int Kx,
    const half_t* __restrict__ Wih, const half_t* __restrict__ Whh,
    const float* __restrict__ bih, const float* __restrict__ bhh,
    const float* __restrict__ lw,  const float* __restrict__ lb,
    float* __restrict__ gates,
    float* __restrict__ h32, half_t* __restrict__ hbf, float* __restrict__ c32,
    float* __restrict__ u, float* __restrict__ du, int* __restrict__ skip,
    float* __restrict__ outF,     // f32 sequence out (layer 1) or nullptr
    half_t* __restrict__ outH,    // f16 sequence out (layer 0) or nullptr
    unsigned* __restrict__ barCnt, unsigned* __restrict__ barGen) {
  const int lane  = threadIdx.x & 31;
  const int gwave = blockIdx.x * (GEMM_THREADS / 32) + (threadIdx.x >> 5); // 0..511
  const int mTile = gwave >> 5;                       // 0..15
  const int nBase = (gwave & 31) * 32;                // 0..992
  const int lr    = lane & 15;
  const int khalf = (lane >> 4) * 8;
  const int mBase = mTile * 16 + 8 * (lane >> 4);

  // Per-wave weight row pointers (L2-resident for the whole layer)
  const half_t* b0 = Wih + (size_t)(nBase + lr)      * Kx;
  const half_t* b1 = Wih + (size_t)(nBase + 16 + lr) * Kx;
  const half_t* c0 = Whh + (size_t)(nBase + lr)      * HC;
  const half_t* c1 = Whh + (size_t)(nBase + 16 + lr) * HC;
  const half_t* hRow = hbf + (size_t)(mTile * 16 + lr) * HC;

  const unsigned nb = gridDim.x;

  for (int t = 0; t < T_; ++t) {
    // ---------------- phase 1: gates GEMM ----------------
    const half_t* aRow =
        Xall + (size_t)t * BS * Kx + (size_t)(mTile * 16 + lr) * Kx;
    if (t + 1 < T_) {
      // pull next timestep's A row toward L2/L0 while this step computes
      __builtin_prefetch(aRow + (size_t)BS * Kx, 0, 2);
      __builtin_prefetch(aRow + (size_t)BS * Kx + 64, 0, 2);
    }

    v8f acc0 = {}; v8f acc1 = {};
    for (int k = 0; k < Kx; k += 32) {
      v16h a  = load_frag(aRow, k, khalf);
      v16h f0 = load_frag(b0,   k, khalf);
      v16h f1 = load_frag(b1,   k, khalf);
      acc0 = __builtin_amdgcn_wmma_f32_16x16x32_f16(false, a, false, f0, (short)0, acc0, false, false);
      acc1 = __builtin_amdgcn_wmma_f32_16x16x32_f16(false, a, false, f1, (short)0, acc1, false, false);
    }
    for (int k = 0; k < HC; k += 32) {
      v16h a  = load_frag(hRow, k, khalf);
      v16h f0 = load_frag(c0,   k, khalf);
      v16h f1 = load_frag(c1,   k, khalf);
      acc0 = __builtin_amdgcn_wmma_f32_16x16x32_f16(false, a, false, f0, (short)0, acc0, false, false);
      acc1 = __builtin_amdgcn_wmma_f32_16x16x32_f16(false, a, false, f1, (short)0, acc1, false, false);
    }
#pragma unroll
    for (int r = 0; r < 8; ++r) {
      gates[(size_t)(mBase + r) * G + nBase + lr]      = acc0[r];
      gates[(size_t)(mBase + r) * G + nBase + 16 + lr] = acc1[r];
    }

    grid_barrier(barCnt, barGen, nb);   // gates complete & visible

    // ---------------- phase 2: skip-LSTM cell ----------------
    if (gwave < BS) {
      const int b = gwave;
      const float ub  = u[b];
      const float dub = du[b];
      const int   skb = skip[b];
      const float bu  = rintf(ub);      // STE round (half-to-even = jnp.round)
      const float omb = 1.0f - bu;

      float nh_e[8], nc_e[8];
      float part = 0.0f;
#pragma unroll
      for (int j = 0; j < 8; ++j) {
        const int idx = lane + j * 32;
        const float gi = gates[(size_t)b * G + idx]       + bih[idx]       + bhh[idx];
        const float gf = gates[(size_t)b * G + 256 + idx] + bih[256 + idx] + bhh[256 + idx];
        const float gg = gates[(size_t)b * G + 512 + idx] + bih[512 + idx] + bhh[512 + idx];
        const float go = gates[(size_t)b * G + 768 + idx] + bih[768 + idx] + bhh[768 + idx];
        const float c  = c32[(size_t)b * HC + idx];
        const float h  = h32[(size_t)b * HC + idx];
        const float cc = sigf(gf) * c + sigf(gi) * tanhf(gg);
        const float ch = sigf(go) * tanhf(cc);
        const float nc_n = cc * bu;
        const float nh_n = ch * bu;
        part += nc_n * lw[idx];
        nh_e[j] = skb ? h * omb : nh_n;
        nc_e[j] = skb ? c * omb : nc_n;
      }
      // wave32 reduction of nc_n . lw
#pragma unroll
      for (int s = 16; s > 0; s >>= 1) part += __shfl_xor(part, s, 32);

      const float du_n = sigf(part + lb[0]);
      const float nu_n = du_n * bu;
      const float nu_s = fminf(fmaxf(ub + dub, 0.0f), 1.0f) * omb;
      const float nu   = skb ? nu_s : nu_n;
      const float ndu  = skb ? dub  : du_n;
      const float n_skips = ceilf(0.5f / nu) - 1.0f;   // nu==0 -> inf -> skip
      const int   nskip = (n_skips > 0.0f) ? 1 : 0;

      float*  oF = outF ? outF + (size_t)t * BS * HC : nullptr;
      half_t* oH = outH ? outH + (size_t)t * BS * HC : nullptr;
#pragma unroll
      for (int j = 0; j < 8; ++j) {
        const int idx = lane + j * 32;
        const size_t o = (size_t)b * HC + idx;
        h32[o] = nh_e[j];
        hbf[o] = (half_t)nh_e[j];
        c32[o] = nc_e[j];
        if (oF) oF[o] = nh_e[j];
        if (oH) oH[o] = (half_t)nh_e[j];
      }
      if (lane == 0) { u[b] = nu; du[b] = ndu; skip[b] = nskip; }
    }

    grid_barrier(barCnt, barGen, nb);   // new h visible before next GEMM
  }
}

// ---------------- plain f32 copy ----------------
__global__ __launch_bounds__(256) void copy_f32(const float* __restrict__ src,
                                                float* __restrict__ dst, int n) {
  int i = blockIdx.x * blockDim.x + threadIdx.x;
  if (i < n) dst[i] = src[i];
}

// ---------------- host orchestration ----------------
extern "C" void kernel_launch(void* const* d_in, const int* in_sizes, int n_in,
                              void* d_out, int out_size, void* d_ws, size_t ws_size,
                              hipStream_t stream) {
  (void)in_sizes; (void)n_in; (void)out_size; (void)ws_size;

  const float* x    = (const float*)d_in[0];
  const float* Wih0 = (const float*)d_in[1];
  const float* Whh0 = (const float*)d_in[2];
  const float* bih0 = (const float*)d_in[3];
  const float* bhh0 = (const float*)d_in[4];
  const float* lw0  = (const float*)d_in[5];
  const float* lb0  = (const float*)d_in[6];
  const float* Wih1 = (const float*)d_in[7];
  const float* Whh1 = (const float*)d_in[8];
  const float* bih1 = (const float*)d_in[9];
  const float* bhh1 = (const float*)d_in[10];
  const float* lw1  = (const float*)d_in[11];
  const float* lb1  = (const float*)d_in[12];

  float* out = (float*)d_out;                       // [T,BS,HC] + hs[2,BS,HC] + cs[2,BS,HC]
  float* hs  = out + (size_t)T_ * BS * HC;
  float* cs  = hs  + (size_t)2  * BS * HC;

  // ---- workspace carve-up (256B aligned) ----
  char* ws = (char*)d_ws;
  size_t off = 0;
  auto carve = [&](size_t bytes) -> char* {
    char* p = ws + off;
    off += (bytes + 255) & ~(size_t)255;
    return p;
  };
  half_t*   xbf    = (half_t*)carve((size_t)T_ * BS * IC * sizeof(half_t));
  half_t*   seq0   = (half_t*)carve((size_t)T_ * BS * HC * sizeof(half_t));
  half_t*   wih0h  = (half_t*)carve((size_t)G * IC * sizeof(half_t));
  half_t*   whh0h  = (half_t*)carve((size_t)G * HC * sizeof(half_t));
  half_t*   wih1h  = (half_t*)carve((size_t)G * HC * sizeof(half_t));
  half_t*   whh1h  = (half_t*)carve((size_t)G * HC * sizeof(half_t));
  float*    gates  = (float*)carve((size_t)BS * G * sizeof(float));
  float*    h32    = (float*)carve((size_t)BS * HC * sizeof(float));
  half_t*   hbf    = (half_t*)carve((size_t)BS * HC * sizeof(half_t));
  float*    c32    = (float*)carve((size_t)BS * HC * sizeof(float));
  float*    uSt    = (float*)carve((size_t)BS * sizeof(float));
  float*    duSt   = (float*)carve((size_t)BS * sizeof(float));
  int*      skipSt = (int*)carve((size_t)BS * sizeof(int));
  unsigned* barCnt = (unsigned*)carve(sizeof(unsigned));
  unsigned* barGen = (unsigned*)carve(sizeof(unsigned));

  auto cvt = [&](const float* s, half_t* d, int n) {
    cvt_f32_f16<<<(n + 255) / 256, 256, 0, stream>>>(s, d, n);
  };
  cvt(x,    xbf,   T_ * BS * IC);
  cvt(Wih0, wih0h, G * IC);
  cvt(Whh0, whh0h, G * HC);
  cvt(Wih1, wih1h, G * HC);
  cvt(Whh1, whh1h, G * HC);

  for (int layer = 0; layer < 2; ++layer) {
    // reset h,c,u each layer; skip/du carry across layers; barrier vars reset
    // only on layer 0 (ws is poisoned before the first timed call)
    init_state<<<(BS * HC + 255) / 256, 256, 0, stream>>>(
        h32, hbf, c32, uSt, duSt, skipSt, barCnt, barGen, layer == 0 ? 1 : 0);

    const half_t* Xall = (layer == 0) ? xbf : seq0;
    const int     Kx   = (layer == 0) ? IC : HC;
    const half_t* Wih  = (layer == 0) ? wih0h : wih1h;
    const half_t* Whh  = (layer == 0) ? whh0h : whh1h;
    const float*  bih  = (layer == 0) ? bih0 : bih1;
    const float*  bhh  = (layer == 0) ? bhh0 : bhh1;
    const float*  lw   = (layer == 0) ? lw0 : lw1;
    const float*  lb   = (layer == 0) ? lb0 : lb1;
    float*        outF = (layer == 1) ? out : nullptr;
    half_t*       outH = (layer == 0) ? seq0 : nullptr;

    skiplstm_layer<<<GEMM_BLOCKS, GEMM_THREADS, 0, stream>>>(
        Xall, Kx, Wih, Whh, bih, bhh, lw, lb, gates,
        h32, hbf, c32, uSt, duSt, skipSt, outF, outH, barCnt, barGen);

    // final h,c of this layer -> hs/cs regions of d_out
    copy_f32<<<(BS * HC + 255) / 256, 256, 0, stream>>>(
        h32, hs + (size_t)layer * BS * HC, BS * HC);
    copy_f32<<<(BS * HC + 255) / 256, 256, 0, stream>>>(
        c32, cs + (size_t)layer * BS * HC, BS * HC);
  }
}
